// WQ_MLP_84799834293037
// MI455X (gfx1250) — compile-verified, matrix operand order checked by reading, not run on hardware
//
#include <hip/hip_runtime.h>

// ---------------------------------------------------------------------------
// MI455X (gfx1250) VQ-MLP forward.
// Roofline: 88 GFLOP over ~0.55 GB moved -> bandwidth floor ~24us @ 23.3 TB/s.
// bf16 WMMA (16x16x32, f32 accum) gives 8x MACs/instr vs f32 16x16x4, moving
// both big GEMMs from compute-bound to the HBM roofline. VQ/BN are noise.
// GEMM staging is fully vectorized and branch-free: b128 global loads from
// clamped addresses, packed b128 LDS stores, uniform-scc tail masking, and a
// double-buffered LDS pipeline (one barrier per 8 WMMAs per wave).
// ---------------------------------------------------------------------------

typedef __attribute__((ext_vector_type(16))) __bf16 v16bf;
typedef __attribute__((ext_vector_type(8)))  float  v8f;

static __device__ __forceinline__ unsigned short f2bf(float x) {
  union { float f; unsigned int u; } v; v.f = x;
  unsigned int r = v.u + 0x7FFFu + ((v.u >> 16) & 1u);  // round-to-nearest-even
  return (unsigned short)(r >> 16);
}
static __device__ __forceinline__ float bf2f(unsigned short b) {
  union { unsigned int u; float f; } v; v.u = ((unsigned int)b) << 16;
  return v.f;
}
static __device__ __forceinline__ unsigned int pk2(float lo, float hi) {
  return (unsigned int)f2bf(lo) | ((unsigned int)f2bf(hi) << 16);
}

// Load a 16-half WMMA fragment from an LDS row laid out [*, 32] (K-major).
// ISA layout (16-bit A 16x32 / B 32x16): lane<16 holds K {koff..koff+7} in
// h[0..7] and K {16+koff..16+koff+7} in h[8..15]; koff = (lane>=16)*8.
static __device__ __forceinline__ v16bf ld_frag(const unsigned short* row, int koff) {
  const __bf16* p = (const __bf16*)row;
  v16bf f;
#pragma unroll
  for (int j = 0; j < 8; ++j) f[j] = p[koff + j];
#pragma unroll
  for (int j = 0; j < 8; ++j) f[8 + j] = p[16 + koff + j];
  return f;
}

// ---------------------------------------------------------------------------
// init: zero BN accumulators and the diff scalar slot of d_out
// ---------------------------------------------------------------------------
__global__ void init_k(float* sums, float* sumsq, float* diffp) {
  int i = threadIdx.x;
  if (i < 512) { sums[i] = 0.f; sumsq[i] = 0.f; }
  if (i == 0) *diffp = 0.f;
}

// ---------------------------------------------------------------------------
// VQ1: W1 (512x784), blocks (out_g=128, in_g=112) of dim 28 (im*4+om), K=512.
// One thread per block; writes qW1 as bf16; accumulates diff1.
// ---------------------------------------------------------------------------
__global__ __launch_bounds__(256) void vq1_k(const float* __restrict__ W1,
                                             const float* __restrict__ E1,
                                             unsigned short* __restrict__ qW1,
                                             float* diffp) {
  const int id = blockIdx.x * 256 + threadIdx.x;  // < 14336
  float bestd = 0.f;
  if (id < 14336) {
    const int og = id / 112, ig = id % 112;
    float w[28];
#pragma unroll
    for (int im = 0; im < 7; ++im)
#pragma unroll
      for (int om = 0; om < 4; ++om)
        w[im * 4 + om] = W1[(og * 4 + om) * 784 + ig * 7 + im];
    bestd = 3.4e38f; int bestk = 0;
    for (int k = 0; k < 512; ++k) {
      float d = 0.f;
#pragma unroll
      for (int dd = 0; dd < 28; ++dd) {
        float e = E1[dd * 512 + k] - w[dd];
        d = fmaf(e, e, d);
      }
      if (d < bestd) { bestd = d; bestk = k; }  // strict < : first-min (argmin)
    }
#pragma unroll
    for (int im = 0; im < 7; ++im)
#pragma unroll
      for (int om = 0; om < 4; ++om)
        qW1[(og * 4 + om) * 784 + ig * 7 + im] = f2bf(E1[(im * 4 + om) * 512 + bestk]);
  }
  __shared__ float red[256];
  red[threadIdx.x] = bestd;
  __syncthreads();
  for (int s = 128; s > 0; s >>= 1) {
    if (threadIdx.x < s) red[threadIdx.x] += red[threadIdx.x + s];
    __syncthreads();
  }
  if (threadIdx.x == 0) atomicAdd(diffp, red[0] * (1.0f / 401408.0f));
}

// ---------------------------------------------------------------------------
// VQ2: W2 (512x512), blocks (out_g=512, in_g=32) of dim 16 (d=im), K=512.
// ---------------------------------------------------------------------------
__global__ __launch_bounds__(256) void vq2_k(const float* __restrict__ W2,
                                             const float* __restrict__ E2,
                                             unsigned short* __restrict__ qW2,
                                             float* diffp) {
  const int id = blockIdx.x * 256 + threadIdx.x;  // < 16384 (exact)
  const int og = id >> 5, ig = id & 31;
  float w[16];
#pragma unroll
  for (int im = 0; im < 16; ++im) w[im] = W2[og * 512 + ig * 16 + im];
  float bestd = 3.4e38f; int bestk = 0;
  for (int k = 0; k < 512; ++k) {
    float d = 0.f;
#pragma unroll
    for (int dd = 0; dd < 16; ++dd) {
      float e = E2[dd * 512 + k] - w[dd];
      d = fmaf(e, e, d);
    }
    if (d < bestd) { bestd = d; bestk = k; }
  }
#pragma unroll
  for (int im = 0; im < 16; ++im)
    qW2[og * 512 + ig * 16 + im] = f2bf(E2[im * 512 + bestk]);
  __shared__ float red[256];
  red[threadIdx.x] = bestd;
  __syncthreads();
  for (int s = 128; s > 0; s >>= 1) {
    if (threadIdx.x < s) red[threadIdx.x] += red[threadIdx.x + s];
    __syncthreads();
  }
  if (threadIdx.x == 0) atomicAdd(diffp, red[0] * (1.0f / 262144.0f));
}

// ---------------------------------------------------------------------------
// WMMA GEMM: Out[m][n] = relu( sum_k A[m][k] * Bw[n][k] + bias[n] ), bf16 out.
//  - A_IS_F32=true : A is f32 (x), converted to bf16 during LDS staging.
//  - A_IS_F32=false: A is bf16 (h), BatchNorm (scale/shift per k) fused.
// Tile: 128x128xK, BK=32, 256 threads = 8 waves, each wave 32x64 (2x4 WMMA).
// Staging is branch-free: clamped global addresses (always in-bounds) and a
// uniform-scc "full tile" flag selecting plain vs tail-masked LDS stores.
// ---------------------------------------------------------------------------
template <bool A_IS_F32>
__global__ __launch_bounds__(256) void gemm_rrt_bf16(
    const void* __restrict__ Av, const unsigned short* __restrict__ Bw,
    const float* __restrict__ bias, const float* __restrict__ scale,
    const float* __restrict__ shift, unsigned short* __restrict__ Out,
    int K, int Nld) {
  __shared__ __align__(16) unsigned short sA[2][128 * 32];  // [m][k] bf16
  __shared__ __align__(16) unsigned short sB[2][128 * 32];  // [n][k] bf16

  const float* Af = (const float*)Av;
  const unsigned short* Ah = (const unsigned short*)Av;

  const int t = threadIdx.x;
  const int m0 = blockIdx.y * 128;
  const int n0 = blockIdx.x * 128;

  // staging: 2 threads per row, one 16-wide K chunk each (16B-aligned)
  const int srow = t >> 1;
  const int scb = (t & 1) * 16;
  const long aRow = (long)(m0 + srow) * K;
  const long bRow = (long)(n0 + srow) * K;

  // wave tiling: waves 0..7 -> (wm in {0,32,64,96}, wn in {0,64})
  const int wave = t >> 5;
  const int wm = (wave >> 1) * 32;
  const int wn = (wave & 1) * 64;
  const int lane = t & 31;
  const int laneLo = lane & 15;
  const int laneHi = lane >> 4;
  const int koff = laneHi * 8;

  const v8f vzero = {0.f, 0.f, 0.f, 0.f, 0.f, 0.f, 0.f, 0.f};
  v8f acc[2][4];
#pragma unroll
  for (int i = 0; i < 2; ++i)
#pragma unroll
    for (int j = 0; j < 4; ++j) acc[i][j] = vzero;

  const int NT = (K + 31) >> 5;

  // staging registers (tile in flight)
  float4 a4[4];
  uint4 ah[2];
  uint4 b4[2];
  float4 s4[4], h4[4];
  bool av = true;  // this thread's 16-chunk is in-bounds

  auto LOAD_TILE = [&](int kt) {
    const int kc = (kt << 5) + scb;
    av = kc < K;                     // K % 16 == 0 -> uniform per 16-chunk
    const int ks = av ? kc : K - 16; // clamped: always in-bounds
    if (A_IS_F32) {
      const float4* p = (const float4*)(Af + aRow + ks);
      a4[0] = p[0]; a4[1] = p[1]; a4[2] = p[2]; a4[3] = p[3];
    } else {
      const uint4* p = (const uint4*)(Ah + aRow + ks);
      ah[0] = p[0]; ah[1] = p[1];
      const float4* ps = (const float4*)(scale + ks);
      const float4* ph = (const float4*)(shift + ks);
      s4[0] = ps[0]; s4[1] = ps[1]; s4[2] = ps[2]; s4[3] = ps[3];
      h4[0] = ph[0]; h4[1] = ph[1]; h4[2] = ph[2]; h4[3] = ph[3];
    }
    const uint4* q = (const uint4*)(Bw + bRow + ks);
    b4[0] = q[0]; b4[1] = q[1];
    if (((kt + 1) << 5) < K) {       // uniform guard -> global_prefetch_b8
      const int kp = (kc + 32 < K) ? kc + 32 : K - 16;  // clamped address
      __builtin_prefetch(Bw + bRow + kp, 0, 1);
      if (A_IS_F32) __builtin_prefetch(Af + aRow + kp, 0, 1);
      else          __builtin_prefetch(Ah + aRow + kp, 0, 1);
    }
  };

  auto STORE_TILE = [&](int buf, bool full /* uniform */) {
    uint4* dA = (uint4*)&sA[buf][srow * 32 + scb];
    uint4* dB = (uint4*)&sB[buf][srow * 32 + scb];
    unsigned int pk[8];
    if (A_IS_F32) {
      const float* af = (const float*)a4;
#pragma unroll
      for (int c = 0; c < 8; ++c) pk[c] = pk2(af[2 * c], af[2 * c + 1]);
    } else {
      const unsigned int* up = (const unsigned int*)ah;
      const float* scf = (const float*)s4;
      const float* shf = (const float*)h4;
#pragma unroll
      for (int c = 0; c < 8; ++c) {
        const unsigned int u = up[c];
        const float lo = fmaf(__uint_as_float(u << 16), scf[2 * c], shf[2 * c]);
        const float hi = fmaf(__uint_as_float(u & 0xFFFF0000u), scf[2 * c + 1], shf[2 * c + 1]);
        pk[c] = pk2(lo, hi);
      }
    }
    uint4 bb0 = b4[0], bb1 = b4[1];
    if (!full) {  // uniform scc branch; per-thread mask kills the OOB tail
      const unsigned int mask = av ? 0xFFFFFFFFu : 0u;
#pragma unroll
      for (int c = 0; c < 8; ++c) pk[c] &= mask;
      bb0.x &= mask; bb0.y &= mask; bb0.z &= mask; bb0.w &= mask;
      bb1.x &= mask; bb1.y &= mask; bb1.z &= mask; bb1.w &= mask;
    }
    dA[0] = make_uint4(pk[0], pk[1], pk[2], pk[3]);
    dA[1] = make_uint4(pk[4], pk[5], pk[6], pk[7]);
    dB[0] = bb0;
    dB[1] = bb1;
  };

  auto COMPUTE_TILE = [&](int buf) {
    const unsigned short* bA = sA[buf];
    const unsigned short* bB = sB[buf];
    v16bf afrag[2], bfrag[4];
#pragma unroll
    for (int i = 0; i < 2; ++i)
      afrag[i] = ld_frag(&bA[(wm + i * 16 + laneLo) * 32], koff);
#pragma unroll
    for (int j = 0; j < 4; ++j)
      bfrag[j] = ld_frag(&bB[(wn + j * 16 + laneLo) * 32], koff);
#pragma unroll
    for (int i = 0; i < 2; ++i)
#pragma unroll
      for (int j = 0; j < 4; ++j)
        acc[i][j] = __builtin_amdgcn_wmma_f32_16x16x32_bf16(
            false, afrag[i], false, bfrag[j], (short)0, acc[i][j], false, false);
  };

  // software pipeline: global loads for tile kt+1 issue before the WMMAs of
  // tile kt; LDS stores go to the alternate buffer. One barrier per tile.
  LOAD_TILE(0);
  STORE_TILE(0, 32 <= K);
  for (int kt = 0; kt < NT; ++kt) {
    const int buf = kt & 1;
    __syncthreads();
    const bool more = (kt + 1) < NT;  // uniform
    if (more) LOAD_TILE(kt + 1);
    COMPUTE_TILE(buf);
    if (more) STORE_TILE(buf ^ 1, ((kt + 2) << 5) <= K);
  }

  // ---- epilogue: +bias, ReLU, store bf16 ----
  // C/D layout: lane<16 -> N=laneLo, M=r ; lane>=16 -> N=laneLo, M=r+8
#pragma unroll
  for (int i = 0; i < 2; ++i) {
#pragma unroll
    for (int j = 0; j < 4; ++j) {
      const int n = n0 + wn + j * 16 + laneLo;
      const float bv = bias[n];
      const long mbase = (long)(m0 + wm + i * 16 + laneHi * 8);
#pragma unroll
      for (int r = 0; r < 8; ++r) {
        float v = acc[i][j][r] + bv;
        v = v > 0.f ? v : 0.f;
        Out[(mbase + r) * Nld + n] = f2bf(v);
      }
    }
  }
}

// ---------------------------------------------------------------------------
// BN column stats over h (65536 x 512 bf16): per-column sum & sumsq (atomics)
// ---------------------------------------------------------------------------
__global__ __launch_bounds__(256) void bn_stats_k(const unsigned short* __restrict__ h,
                                                  float* sums, float* sumsq) {
  const int col = blockIdx.x * 256 + threadIdx.x;  // grid.x = 2 -> 512 cols
  const long mbase = (long)blockIdx.y * 1024;      // grid.y = 64
  float s = 0.f, ss = 0.f;
  for (int r = 0; r < 1024; ++r) {
    const float v = bf2f(h[(mbase + r) * 512 + col]);
    s += v;
    ss = fmaf(v, v, ss);
  }
  atomicAdd(&sums[col], s);
  atomicAdd(&sumsq[col], ss);
}

__global__ void bn_final_k(const float* sums, const float* sumsq,
                           const float* gamma, const float* beta,
                           float* scale, float* shift) {
  const int c = blockIdx.x * 256 + threadIdx.x;
  if (c >= 512) return;
  const float mu = sums[c] * (1.f / 65536.f);
  const float var = sumsq[c] * (1.f / 65536.f) - mu * mu;  // biased var
  const float rs = rsqrtf(var + 1e-5f);
  const float sc = gamma[c] * rs;
  scale[c] = sc;
  shift[c] = beta[c] - mu * sc;
}

// ---------------------------------------------------------------------------
// GEMM3: out[m][j] = sum_k h2[m][k]*W3[j][k] + b3[j]; N=10 (memory-trivial).
// 256 threads = 16 rows x 16 cols (6 idle cols); W3 staged in LDS.
// ---------------------------------------------------------------------------
__global__ __launch_bounds__(256) void gemm3_k(const unsigned short* __restrict__ h2,
                                               const float* __restrict__ W3,
                                               const float* __restrict__ b3,
                                               float* __restrict__ out) {
  __shared__ float sW[10 * 512];
  __shared__ float sb[16];
  const int t = threadIdx.x;
  for (int i = t; i < 10 * 512; i += 256) sW[i] = W3[i];
  if (t < 16) sb[t] = (t < 10) ? b3[t] : 0.f;
  __syncthreads();

  const int j = t & 15;
  const long m = (long)blockIdx.x * 16 + (t >> 4);
  float acc = 0.f;
  const unsigned short* hrow = h2 + m * 512;
  for (int k = 0; k < 512; ++k) {
    const float hv = bf2f(hrow[k]);  // broadcast across the 16 j-lanes
    acc = fmaf(hv, (j < 10) ? sW[j * 512 + k] : 0.f, acc);
  }
  if (j < 10) out[m * 10 + j] = acc + sb[j];
}

// ---------------------------------------------------------------------------
// launch
// ---------------------------------------------------------------------------
extern "C" void kernel_launch(void* const* d_in, const int* in_sizes, int n_in,
                              void* d_out, int out_size, void* d_ws, size_t ws_size,
                              hipStream_t stream) {
  const float* x      = (const float*)d_in[0];
  const float* W1     = (const float*)d_in[1];
  const float* b1     = (const float*)d_in[2];
  const float* gamma1 = (const float*)d_in[3];
  const float* beta1  = (const float*)d_in[4];
  const float* E1     = (const float*)d_in[5];
  const float* W2     = (const float*)d_in[6];
  const float* b2     = (const float*)d_in[7];
  const float* E2     = (const float*)d_in[8];
  const float* W3     = (const float*)d_in[9];
  const float* b3     = (const float*)d_in[10];

  float* out = (float*)d_out;
  float* diffp = out + (size_t)65536 * 10;  // diff1+diff2 scalar slot

  // workspace layout (bytes, 256-aligned)
  char* ws = (char*)d_ws;
  unsigned short* qW1 = (unsigned short*)(ws + 0);          //   802,816 B
  unsigned short* qW2 = (unsigned short*)(ws + 802816);     //   524,288 B
  unsigned short* h   = (unsigned short*)(ws + 1327104);    // 67,108,864 B
  unsigned short* h2  = (unsigned short*)(ws + 68435968);   // 67,108,864 B
  float* sums  = (float*)(ws + 135544832);                  // 512 f32
  float* sumsq = sums + 512;
  float* scale = sums + 1024;
  float* shift = sums + 1536;

  init_k<<<1, 512, 0, stream>>>(sums, sumsq, diffp);
  vq1_k<<<56, 256, 0, stream>>>(W1, E1, qW1, diffp);
  vq2_k<<<64, 256, 0, stream>>>(W2, E2, qW2, diffp);

  // h = relu(x @ qW1^T + b1), bf16
  gemm_rrt_bf16<true><<<dim3(4, 512), 256, 0, stream>>>(
      (const void*)x, qW1, b1, nullptr, nullptr, h, 784, 512);

  bn_stats_k<<<dim3(2, 64), 256, 0, stream>>>(h, sums, sumsq);
  bn_final_k<<<2, 256, 0, stream>>>(sums, sumsq, gamma1, beta1, scale, shift);

  // h2 = relu(bn(h) @ qW2^T + b2), BN fused into A-staging
  gemm_rrt_bf16<false><<<dim3(4, 512), 256, 0, stream>>>(
      (const void*)h, qW2, b2, scale, shift, h2, 512, 512);

  gemm3_k<<<4096, 256, 0, stream>>>(h2, W3, b3, out);
}
